// AdaDConv_29429115912590
// MI455X (gfx1250) — compile-verified, matrix operand order checked by reading, not run on hardware
//
#include <hip/hip_runtime.h>
#include <hip/hip_bf16.h>
#include <math.h>

typedef __attribute__((ext_vector_type(16))) _Float16 v16h;
typedef __attribute__((ext_vector_type(8)))  float    v8f;

#define B_   8
#define C_   256
#define H_   128
#define W_   128
#define OH_  64
#define OW_  64
#define K2_  9
#define EPS_ 1e-5f
#define CPL  (H_ * W_)            // 16384 elems per (b,c) plane

// ---------------------------------------------------------------------------
// 1) Global average pool: gap[b*C+c] = mean(x[b,c,:,:])
// ---------------------------------------------------------------------------
__global__ void gap_kernel(const float* __restrict__ x, float* __restrict__ gap) {
    const int bc = blockIdx.x;                       // 0..2047
    const float4* p = (const float4*)(x + (size_t)bc * CPL);
    float s = 0.f;
    for (int idx = threadIdx.x; idx < CPL / 4; idx += 256) {
        float4 v = p[idx];
        s += v.x + v.y + v.z + v.w;
    }
    for (int off = 16; off > 0; off >>= 1) s += __shfl_down(s, off, 32);
    __shared__ float red[8];
    if ((threadIdx.x & 31) == 0) red[threadIdx.x >> 5] = s;
    __syncthreads();
    if (threadIdx.x == 0) {
        float t = 0.f;
        #pragma unroll
        for (int w = 0; w < 8; ++w) t += red[w];
        gap[bc] = t * (1.f / (float)CPL);
    }
}

// ---------------------------------------------------------------------------
// 2) channel net: ch = relu(gap @ W1^T) @ W2^T      W1:(64,256)  W2:(256,64)
// ---------------------------------------------------------------------------
__global__ void chnet_kernel(const float* __restrict__ gap,
                             const float* __restrict__ w1,
                             const float* __restrict__ w2,
                             float* __restrict__ ch) {
    __shared__ float hidden[B_][64];
    const int tid = threadIdx.x;
    for (int idx = tid; idx < B_ * 64; idx += 256) {
        const int b = idx >> 6, h = idx & 63;
        float s = 0.f;
        const float* g = gap + b * C_;
        const float* w = w1 + h * C_;
        for (int c = 0; c < C_; ++c) s += g[c] * w[c];
        hidden[b][h] = fmaxf(s, 0.f);
    }
    __syncthreads();
    for (int idx = tid; idx < B_ * C_; idx += 256) {
        const int b = idx >> 8, c = idx & 255;
        float s = 0.f;
        const float* w = w2 + c * 64;
        #pragma unroll 8
        for (int h = 0; h < 64; ++h) s += hidden[b][h] * w[h];
        ch[idx] = s;
    }
}

// ---------------------------------------------------------------------------
// 3) Pack conv weights into WMMA 16x32 f16 A-fragment layout.
//    step = tap*8 + cc  (tap = ki*3+kj, cc = channel chunk of 32)
//    lane = g*16 + m (m = output tap index, rows >= 9 are zero)
//    element e -> K = (e<8 ? e : e+8) + 8*g ;  c = cc*32 + K
// ---------------------------------------------------------------------------
__global__ void pack_kernel(const float* __restrict__ w_conv,
                            _Float16* __restrict__ packedA) {
    const int idx = blockIdx.x * 256 + threadIdx.x;   // 72*32*16 = 36864 total
    if (idx >= 72 * 32 * 16) return;
    const int step = idx >> 9;            // /512
    const int rem  = idx & 511;
    const int lane = rem >> 4;
    const int e    = rem & 15;
    const int g    = lane >> 4;
    const int m    = lane & 15;           // tap-output row
    const int K    = (e < 8 ? e : e + 8) + g * 8;
    const int cc   = step & 7;
    const int tap  = step >> 3;
    const int c    = cc * 32 + K;
    const int ki   = tap / 3, kj = tap % 3;
    float v = 0.f;
    if (m < K2_) v = w_conv[(((size_t)m * C_ + c) * 3 + ki) * 3 + kj];
    packedA[idx] = (_Float16)v;
}

// ---------------------------------------------------------------------------
// 4) Conv (zero-padded, stride 2) + BN via WMMA f32_16x16x32_f16.
//    One block = (b, output row i). 4 waves x 16 pixels = 64 output pixels.
//    LDS slot permutation (swap bits 3<->4 of channel) makes each lane's
//    16-half B-fragment one contiguous 32B run -> single v16h LDS load.
// ---------------------------------------------------------------------------
__global__ void conv_wmma_kernel(const float* __restrict__ x,
                                 const _Float16* __restrict__ packedA,
                                 const float* __restrict__ bn_g,
                                 const float* __restrict__ bn_b,
                                 const float* __restrict__ bn_m,
                                 const float* __restrict__ bn_v,
                                 float* __restrict__ wlog) {
    __shared__ __align__(32) _Float16 xs[3][130][32];
    const int b   = blockIdx.x >> 6;
    const int i   = blockIdx.x & 63;
    const int tid = threadIdx.x;          // 128 threads = 4 waves
    const int lane = tid & 31;
    const int wave = tid >> 5;
    const int gh   = lane >> 4;           // half-group
    const int N    = lane & 15;
    const int j    = wave * 16 + N;       // output column
    const int colb = 2 * j;               // padded-array column base (kj adds 0..2)

    const float* xb = x + (size_t)b * C_ * CPL;
    v8f acc = {};

    for (int cc = 0; cc < 8; ++cc) {
        __syncthreads();
        // stage 32 channels x 3 rows x 130 cols (zero padded) as f16,
        // channel slot-permuted so B-fragments are contiguous per lane.
        for (int idx = tid; idx < 3 * 130 * 32; idx += 128) {
            const int cola = idx % 130;
            const int cl   = (idx / 130) & 31;
            const int ki   = idx / (130 * 32);
            const int slot = (cl & 7) | ((cl & 8) << 1) | ((cl & 16) >> 1);
            const int y    = 2 * i - 1 + ki;
            const int xc   = cola - 1;
            float v = 0.f;
            if (y >= 0 && y < H_ && xc >= 0 && xc < W_) {
                const float* src = xb + (size_t)(cc * 32 + cl) * CPL + y * W_ + xc;
                v = *src;
                if (cc < 7) __builtin_prefetch(src + 32 * CPL, 0, 0);
            }
            xs[ki][cola][slot] = (_Float16)v;
        }
        __syncthreads();

        for (int tap = 0; tap < 9; ++tap) {
            const int ki = tap / 3, kj = tap % 3;
            const int step = tap * 8 + cc;
            // A fragment: 16 f16 per lane, contiguous (2x b128 global loads)
            v16h a = *(const v16h*)(packedA + ((size_t)step * 32 + lane) * 16);
            // B fragment: one contiguous 32B run per lane (2x ds_load_b128)
            v16h bf = *(const v16h*)&xs[ki][colb + kj][gh * 16];
            acc = __builtin_amdgcn_wmma_f32_16x16x32_f16(
                false, a, false, bf, (short)0, acc, false, false);
        }
    }

    // Epilogue: rows M = r + 8*gh are taps; apply eval-mode BatchNorm.
    #pragma unroll
    for (int r = 0; r < 8; ++r) {
        const int t = r + 8 * gh;
        if (t < K2_) {
            const float sc = bn_g[t] * rsqrtf(bn_v[t] + EPS_);
            const float v  = (acc[r] - bn_m[t]) * sc + bn_b[t];
            wlog[(((size_t)b * K2_ + t) * OH_ + i) * OW_ + j] = v;
        }
    }
}

// ---------------------------------------------------------------------------
// 5) softmax over taps * reflect-padded patches, summed over taps.
//    One block = (b, output row i); 256 threads cover 32 channels x 64 cols.
//    x rows staged with GLOBAL_LOAD_ASYNC_TO_LDS_B128 (ASYNCcnt-tracked,
//    no VGPR round trip); reflection folded into source-row index and a
//    read-time left-edge fix, so the copy is pure interior 16B segments.
// ---------------------------------------------------------------------------
__global__ void final_kernel(const float* __restrict__ x,
                             const float* __restrict__ wlog,
                             const float* __restrict__ ch,
                             float* __restrict__ out) {
    __shared__ __align__(16) float xs[3][32][128];   // [ki][c_local][col]
    __shared__ float wl[K2_][OW_];
    const int b   = blockIdx.x >> 6;
    const int i   = blockIdx.x & 63;
    const int tid = threadIdx.x;

    for (int idx = tid; idx < K2_ * OW_; idx += 256) {
        const int t = idx / OW_, j = idx % OW_;
        wl[t][j] = wlog[(((size_t)b * K2_ + t) * OH_ + i) * OW_ + j];
    }

    // reflect-padded source rows for this output row (pad=1 reflect)
    int yr[3];
    #pragma unroll
    for (int ki = 0; ki < 3; ++ki) {
        int y = 2 * i - 1 + ki;
        if (y < 0) y = -y; else if (y > H_ - 1) y = 2 * (H_ - 1) - y;
        yr[ki] = y;
    }

    const int cl = tid >> 3;   // channel within chunk
    const int jg = tid & 7;    // column group
    const float* xb = x + (size_t)b * C_ * CPL;

    for (int cc = 0; cc < 8; ++cc) {
        __syncthreads();
        // async-copy 96 rows x 512B (3 ki x 32 ch x 128 cols) into LDS
        for (int idx = tid; idx < 3 * 32 * 32; idx += 256) {
            const int seg = idx & 31;          // 16B segment within row
            const int row = idx >> 5;          // 0..95
            const int c2  = row & 31;
            const int ki  = row >> 5;
            const unsigned ldsa = (unsigned)(uintptr_t)&xs[ki][c2][seg * 4];
            const unsigned goff =
                (unsigned)((((cc * 32 + c2) * CPL + yr[ki] * W_) + seg * 4) * 4);
            asm volatile("global_load_async_to_lds_b128 %0, %1, %2"
                         :: "v"(ldsa), "v"(goff), "s"(xb) : "memory");
        }
        asm volatile("s_wait_asynccnt 0x0" ::: "memory");
        __syncthreads();

        const int c = cc * 32 + cl;
        const float chv = ch[b * C_ + c];
        float* op = out + (((size_t)b * C_ + c) * OH_ + i) * OW_;
        for (int q = 0; q < 8; ++q) {
            const int j  = jg * 8 + q;
            const int c0 = 2 * j - 1;                 // leftmost tap column
            const int ci0 = (c0 < 0) ? 1 : c0;        // reflect left edge
            float l[K2_], m = -INFINITY;
            #pragma unroll
            for (int t = 0; t < K2_; ++t) {
                l[t] = chv * wl[t][j];
                m = fmaxf(m, l[t]);
            }
            float s = 0.f, v = 0.f;
            #pragma unroll
            for (int t = 0; t < K2_; ++t) {
                const float e = __expf(l[t] - m);
                s += e;
                const int kj = t % 3;
                const int ci = (kj == 0) ? ci0 : (c0 + kj);
                v += e * xs[t / 3][cl][ci];
            }
            op[j] = v / s;
        }
    }
}

// ---------------------------------------------------------------------------
extern "C" void kernel_launch(void* const* d_in, const int* in_sizes, int n_in,
                              void* d_out, int out_size, void* d_ws, size_t ws_size,
                              hipStream_t stream) {
    const float* x       = (const float*)d_in[0];
    const float* w_conv  = (const float*)d_in[1];
    const float* bn_g    = (const float*)d_in[2];
    const float* bn_b    = (const float*)d_in[3];
    const float* bn_m    = (const float*)d_in[4];
    const float* bn_v    = (const float*)d_in[5];
    const float* ch_w1   = (const float*)d_in[6];
    const float* ch_w2   = (const float*)d_in[7];
    float* out = (float*)d_out;

    char* ws = (char*)d_ws;
    float*    gap     = (float*)(ws);                       // 2048 f32
    float*    ch      = (float*)(ws + 8192);                // 2048 f32
    _Float16* packedA = (_Float16*)(ws + 16384);            // 72*32*16 f16
    float*    wlog    = (float*)(ws + 16384 + 90112);       // 8*9*64*64 f32

    gap_kernel  <<<B_ * C_,              256, 0, stream>>>(x, gap);
    chnet_kernel<<<1,                    256, 0, stream>>>(gap, ch_w1, ch_w2, ch);
    pack_kernel <<<(72 * 32 * 16) / 256, 256, 0, stream>>>(w_conv, packedA);
    conv_wmma_kernel<<<B_ * OH_,         128, 0, stream>>>(x, packedA, bn_g, bn_b,
                                                           bn_m, bn_v, wlog);
    final_kernel<<<B_ * OH_,             256, 0, stream>>>(x, wlog, ch, out);
}
